// BiLSTMLayer_83726092468474
// MI455X (gfx1250) — matrix operand order repaired and checked by reading, not compiled, not run on hardware
//
#include <hip/hip_runtime.h>

// ---------------------------------------------------------------------------
// Problem constants (match reference)
// ---------------------------------------------------------------------------
constexpr int kB  = 64;    // batch
constexpr int kT  = 1024;  // time
constexpr int kF  = 128;   // input features
constexpr int kU  = 256;   // LSTM units
constexpr int kNG = 4 * kU;   // 1024 gate columns (i,f,g,o)
constexpr int kH  = 2 * kU;   // 512 BiLSTM feature width
constexpr float kLnEps = 1e-3f;

// ---------------------------------------------------------------------------
// CDNA5 WMMA types
// ---------------------------------------------------------------------------
typedef __attribute__((ext_vector_type(16))) __bf16        v16bf;
typedef __attribute__((ext_vector_type(8)))  float         v8f;
typedef __attribute__((ext_vector_type(4)))  unsigned int  u32x4;

union Frag {
    u32x4 q[2];
    v16bf v;
};

__device__ __forceinline__ float bf2f(unsigned short s) {
    return __uint_as_float((unsigned int)s << 16);
}
__device__ __forceinline__ unsigned short f2bf(float f) {
    unsigned int u = __float_as_uint(f);
    unsigned int r = u + 0x7fffu + ((u >> 16) & 1u);   // round-to-nearest-even
    return (unsigned short)(r >> 16);
}
__device__ __forceinline__ float sigf(float x) {
    return 1.0f / (1.0f + __expf(-x));
}
__device__ __forceinline__ float tanh_fast(float x) {
    return 1.0f - 2.0f / (__expf(2.0f * x) + 1.0f);
}

// ---------------------------------------------------------------------------
// f32 -> bf16 elementwise convert (x sequence)
// ---------------------------------------------------------------------------
__global__ __launch_bounds__(256)
void f32_to_bf16_kernel(const float* __restrict__ src,
                        unsigned short* __restrict__ dst, int n) {
    int i = blockIdx.x * 256 + threadIdx.x;
    if (i < n) dst[i] = f2bf(src[i]);
}

// ---------------------------------------------------------------------------
// Build weight matrix [W;U]^T in WMMA-B-fragment-tiled bf16 layout.
// Fragment (ntile, kb32) is 512 elems: lane L holds col n = ntile*16+(L&15),
// k = kb32*32 + (L>>4)*16 + e (e = 0..15 contiguous) -> per-lane 32B chunk at
// frag_base + L*16 elems. All step-kernel B loads are coalesced b128 pairs.
// ---------------------------------------------------------------------------
__global__ __launch_bounds__(256)
void build_wt_kernel(unsigned short* __restrict__ dst,
                     const float* __restrict__ W,    // [Fin, 1024]
                     const float* __restrict__ Uw,   // [kU, 1024]
                     int Fin, int KTOT) {
    int i = blockIdx.x * 256 + threadIdx.x;
    int total = kNG * KTOT;
    if (i >= total) return;
    int frag   = i >> 9;          // / 512
    int within = i & 511;
    int lane   = within >> 4;
    int e      = within & 15;
    int KB32   = KTOT / 32;
    int ntile  = frag / KB32;
    int kb32   = frag % KB32;
    int n = ntile * 16 + (lane & 15);
    int k = kb32 * 32 + (lane >> 4) * 16 + e;
    float v = (k < Fin) ? W[(size_t)k * kNG + n]
                        : Uw[(size_t)(k - Fin) * kNG + n];
    dst[i] = f2bf(v);
}

// ---------------------------------------------------------------------------
// One BiLSTM timestep (both directions). Grid = 16 blocks x 256 threads:
//   blockIdx: bit3 = dir (0 fwd / 1 bwd), bits2:1 = mtile (batch/16),
//             bit0 = unit-half; wave (8/block) -> utile = half*8 + wave.
// Each wave: 16x16 (batch x unit) tile, 4 gate accumulators, KTOT/32 WMMAs
// per gate with v_wmma_f32_16x16x32_bf16; then gate math + c/h update.
// ---------------------------------------------------------------------------
template <int KTOT, int LAYER>
__global__ __launch_bounds__(256)
void lstm_step_kernel(const unsigned short* __restrict__ xseq,  // [kB,kT,XF] bf16
                      const unsigned short* __restrict__ WTf,   // tiled [kNG*KTOT]
                      const unsigned short* __restrict__ WTb,
                      const float* __restrict__ biasf,          // [4U]
                      const float* __restrict__ biasb,
                      float* __restrict__ cstate,               // [2][kB][kU]
                      unsigned short* __restrict__ hping,       // [2][2][kB][kU]
                      unsigned short* __restrict__ h1out,       // layer1: [kB,kT,512]
                      float* __restrict__ out,                  // layer2: [kB,kT,512]
                      int t) {
    constexpr int XF   = KTOT - kU;   // x-part width (128 or 512)
    constexpr int PAD  = 8;           // row stride = KTOT+8 elems -> 4-bank skew
    constexpr int KB32 = KTOT / 32;
    __shared__ __align__(16) unsigned short Atile[16][KTOT + PAD];

    const int dir   = blockIdx.x >> 3;
    const int mtile = (blockIdx.x >> 1) & 3;
    const int uhalf = blockIdx.x & 1;
    const int wave  = threadIdx.x >> 5;
    const int lane  = threadIdx.x & 31;
    const int utile = uhalf * 8 + wave;
    const int time  = dir ? (kT - 1 - t) : t;

    const unsigned short* WT    = dir ? WTb : WTf;
    const float*          bias  = dir ? biasb : biasf;
    const unsigned short* hprev = hping + (size_t)(dir * 2 + (t & 1)) * kB * kU;
    unsigned short*       hnext = hping + (size_t)(dir * 2 + ((t + 1) & 1)) * kB * kU;

    // ---- stage A = [x_t | h_prev] (16 batch rows x KTOT) into LDS ----------
    constexpr int CH = KTOT / 8;                 // 16B chunks per row
    for (int idx = threadIdx.x; idx < 16 * CH; idx += 256) {
        const int r = idx / CH;
        const int k = (idx % CH) * 8;
        const int b = mtile * 16 + r;
        const unsigned short* src =
            (k < XF) ? (xseq + ((size_t)b * kT + time) * XF + k)
                     : (hprev + (size_t)b * kU + (k - XF));
        *(u32x4*)&Atile[r][k] = *(const u32x4*)src;
    }
    __syncthreads();

    // ---- WMMA accumulation: 4 gates x KB32 k-blocks ------------------------
    v8f zero = {0.f, 0.f, 0.f, 0.f, 0.f, 0.f, 0.f, 0.f};
    v8f acc[4] = {zero, zero, zero, zero};

    const int arow = lane & 15;
    const int akc  = (lane >> 4) * 16;   // A: K chunk base (0 or 16)

    for (int kb32 = 0; kb32 < KB32; ++kb32) {
        const int kb = kb32 * 32;
        Frag a;
        a.q[0] = *(const u32x4*)&Atile[arow][kb + akc];
        a.q[1] = *(const u32x4*)&Atile[arow][kb + akc + 8];
#pragma unroll
        for (int g = 0; g < 4; ++g) {
            const int ntile = g * 16 + utile;
            const unsigned short* bp =
                WT + ((size_t)(ntile * KB32 + kb32) << 9) + lane * 16;
            Frag bfr;
            bfr.q[0] = *(const u32x4*)bp;
            bfr.q[1] = *(const u32x4*)(bp + 8);
            acc[g] = __builtin_amdgcn_wmma_f32_16x16x32_bf16(
                false, a.v, false, bfr.v, (short)0, acc[g], false, false);
        }
    }

    // ---- gate math + cell/hidden update -----------------------------------
    const int unit = utile * 16 + (lane & 15);
    const float bi = bias[0 * kU + unit];
    const float bf = bias[1 * kU + unit];
    const float bg = bias[2 * kU + unit];
    const float bo = bias[3 * kU + unit];

#pragma unroll
    for (int r = 0; r < 8; ++r) {
        const int brow = mtile * 16 + r + ((lane >> 4) << 3);  // C/D layout
        const size_t ci = ((size_t)dir * kB + brow) * kU + unit;
        const float zi = acc[0][r] + bi;
        const float zf = acc[1][r] + bf;
        const float zg = acc[2][r] + bg;
        const float zo = acc[3][r] + bo;
        const float cold = cstate[ci];
        const float cn = sigf(zf) * cold + sigf(zi) * tanh_fast(zg);
        const float h  = sigf(zo) * tanh_fast(cn);
        cstate[ci] = cn;
        const unsigned short hb = f2bf(h);
        hnext[(size_t)brow * kU + unit] = hb;
        if constexpr (LAYER == 1) {
            h1out[((size_t)brow * kT + time) * kH + dir * kU + unit] = hb;
        } else {
            out[((size_t)brow * kT + time) * kH + dir * kU + unit] = h;
        }
    }
}

// ---------------------------------------------------------------------------
// In-place LayerNorm over 512 features; one wave32 per row.
// ---------------------------------------------------------------------------
__global__ __launch_bounds__(256)
void ln_kernel(unsigned short* __restrict__ h1,
               const float* __restrict__ gamma,
               const float* __restrict__ beta) {
    const int row  = (blockIdx.x * 256 + threadIdx.x) >> 5;
    const int lane = threadIdx.x & 31;
    if (row >= kB * kT) return;
    unsigned short* rp = h1 + (size_t)row * kH + lane * 16;
    u32x4 q0 = *(const u32x4*)rp;
    u32x4 q1 = *(const u32x4*)(rp + 8);

    float v[16];
    float s = 0.f, s2 = 0.f;
#pragma unroll
    for (int i = 0; i < 4; ++i) {
        unsigned int w0 = q0[i], w1 = q1[i];
        v[2 * i + 0] = bf2f((unsigned short)(w0 & 0xffffu));
        v[2 * i + 1] = bf2f((unsigned short)(w0 >> 16));
        v[8 + 2 * i + 0] = bf2f((unsigned short)(w1 & 0xffffu));
        v[8 + 2 * i + 1] = bf2f((unsigned short)(w1 >> 16));
    }
#pragma unroll
    for (int e = 0; e < 16; ++e) { s += v[e]; s2 += v[e] * v[e]; }
    for (int off = 16; off > 0; off >>= 1) {
        s  += __shfl_xor(s, off);
        s2 += __shfl_xor(s2, off);
    }
    const float mean = s * (1.0f / kH);
    const float var  = s2 * (1.0f / kH) - mean * mean;
    const float rstd = rsqrtf(var + kLnEps);

    float y[16];
#pragma unroll
    for (int e = 0; e < 16; ++e) {
        const int j = lane * 16 + e;
        y[e] = (v[e] - mean) * rstd * gamma[j] + beta[j];
    }
    u32x4 o0, o1;
#pragma unroll
    for (int i = 0; i < 4; ++i) {
        o0[i] = (unsigned int)f2bf(y[2 * i]) | ((unsigned int)f2bf(y[2 * i + 1]) << 16);
        o1[i] = (unsigned int)f2bf(y[8 + 2 * i]) | ((unsigned int)f2bf(y[8 + 2 * i + 1]) << 16);
    }
    *(u32x4*)rp = o0;
    *(u32x4*)(rp + 8) = o1;
}

// ---------------------------------------------------------------------------
// Host launch
// ---------------------------------------------------------------------------
extern "C" void kernel_launch(void* const* d_in, const int* in_sizes, int n_in,
                              void* d_out, int out_size, void* d_ws, size_t ws_size,
                              hipStream_t stream) {
    (void)in_sizes; (void)n_in; (void)out_size; (void)ws_size;
    const float* x     = (const float*)d_in[0];
    const float* W1f   = (const float*)d_in[1];
    const float* U1f   = (const float*)d_in[2];
    const float* b1f   = (const float*)d_in[3];
    const float* W1b   = (const float*)d_in[4];
    const float* U1b   = (const float*)d_in[5];
    const float* b1b   = (const float*)d_in[6];
    const float* gamma = (const float*)d_in[7];
    const float* beta  = (const float*)d_in[8];
    const float* W2f   = (const float*)d_in[9];
    const float* U2f   = (const float*)d_in[10];
    const float* b2f   = (const float*)d_in[11];
    const float* W2b   = (const float*)d_in[12];
    const float* U2b   = (const float*)d_in[13];
    const float* b2b   = (const float*)d_in[14];

    char* p = (char*)d_ws;
    auto take = [&](size_t bytes) -> char* {
        char* r = p;
        p += (bytes + 255) & ~(size_t)255;
        return r;
    };
    unsigned short* xbf  = (unsigned short*)take((size_t)kB * kT * kF * 2);   // 16 MB
    unsigned short* h1   = (unsigned short*)take((size_t)kB * kT * kH * 2);   // 64 MB
    unsigned short* WT1f = (unsigned short*)take((size_t)kNG * 384 * 2);
    unsigned short* WT1b = (unsigned short*)take((size_t)kNG * 384 * 2);
    unsigned short* WT2f = (unsigned short*)take((size_t)kNG * 768 * 2);
    unsigned short* WT2b = (unsigned short*)take((size_t)kNG * 768 * 2);
    float*          cst1 = (float*)take((size_t)2 * kB * kU * 4);
    float*          cst2 = (float*)take((size_t)2 * kB * kU * 4);
    unsigned short* hp1  = (unsigned short*)take((size_t)4 * kB * kU * 2);
    unsigned short* hp2  = (unsigned short*)take((size_t)4 * kB * kU * 2);

    // --- prep: bf16 x, fragment-tiled bf16 [W;U]^T, zero states -------------
    {
        int n = kB * kT * kF;
        f32_to_bf16_kernel<<<(n + 255) / 256, 256, 0, stream>>>(x, xbf, n);
    }
    build_wt_kernel<<<(kNG * 384 + 255) / 256, 256, 0, stream>>>(WT1f, W1f, U1f, kF, 384);
    build_wt_kernel<<<(kNG * 384 + 255) / 256, 256, 0, stream>>>(WT1b, W1b, U1b, kF, 384);
    build_wt_kernel<<<(kNG * 768 + 255) / 256, 256, 0, stream>>>(WT2f, W2f, U2f, kH, 768);
    build_wt_kernel<<<(kNG * 768 + 255) / 256, 256, 0, stream>>>(WT2b, W2b, U2b, kH, 768);
    hipMemsetAsync(cst1, 0, (size_t)2 * kB * kU * 4, stream);
    hipMemsetAsync(cst2, 0, (size_t)2 * kB * kU * 4, stream);
    hipMemsetAsync(hp1, 0, (size_t)4 * kB * kU * 2, stream);
    hipMemsetAsync(hp2, 0, (size_t)4 * kB * kU * 2, stream);

    // --- layer 1 BiLSTM (fwd + bwd fused per step) --------------------------
    for (int t = 0; t < kT; ++t) {
        lstm_step_kernel<384, 1><<<16, 256, 0, stream>>>(
            xbf, WT1f, WT1b, b1f, b1b, cst1, hp1, h1, nullptr, t);
    }
    // --- LayerNorm (in place on h1) -----------------------------------------
    ln_kernel<<<(kB * kT) / 8, 256, 0, stream>>>(h1, gamma, beta);

    // --- layer 2 BiLSTM -> d_out (f32) --------------------------------------
    for (int t = 0; t < kT; ++t) {
        lstm_step_kernel<768, 2><<<16, 256, 0, stream>>>(
            h1, WT2f, WT2b, b2f, b2b, cst2, hp2, nullptr, (float*)d_out, t);
    }
}
